// Mamba_TransformerEncoderLayer_9165460210046
// MI455X (gfx1250) — matrix operand order repaired and checked by reading, not compiled, not run on hardware
//
#include <hip/hip_runtime.h>
#include <hip/hip_bf16.h>

typedef __attribute__((ext_vector_type(16))) _Float16 v16h;
typedef __attribute__((ext_vector_type(8)))  _Float16 v8h;
typedef __attribute__((ext_vector_type(4)))  _Float16 v4h;
typedef __attribute__((ext_vector_type(8)))  float    v8f;

// ---- problem constants (match reference) ----
constexpr int Bb   = 2;
constexpr int Ls   = 512;
constexpr int Dd   = 1024;
constexpr int Hh   = 16;
constexpr int FFf  = 4096;
constexpr int DIi  = 2 * Dd;      // 2048
constexpr int DSs  = 64;
constexpr int DCc  = 4;
constexpr int DTRr = 64;
constexpr int HDd  = Dd / Hh;     // 64
constexpr int MR   = Bb * Ls;     // 1024 token rows
constexpr float EPSf = 1e-5f;

// =====================================================================
// Generic WMMA GEMM:  C[z][m][n] = act(alpha * sum_k A[m,k]*W[n,k] + bias[n])
//   A element:  A + zb*sAb + zh*sAh + m*ldaM + k          (k always contiguous)
//   W element:  W + zb*sWb + zh*sWh + n*ldwN + k*ldwK     (WVEC => ldwK==1)
//   C element:  C + zb*sCb + zh*sCh + m*ldc  + n
// 4 waves / 128-thread block; each wave owns a 16x64 slab of a 64x64 tile.
// Software-pipelined: global float4 loads for tile k+1 are issued before
// the WMMAs of tile k; LDS stores happen from registers at the top of the
// next iteration. Fragments load as 16B ds_load_b128 pairs (LDS pitch 80B).
// =====================================================================
template <bool WVEC>
__global__ __launch_bounds__(128)
void gemm_wmma_kernel(const float* __restrict__ A, int ldaM,
                      const float* __restrict__ W, int ldwN, int ldwK,
                      const float* __restrict__ bias,
                      float* __restrict__ C, int ldc,
                      int K, int act, float alpha, int zdiv,
                      long long sAb, long long sAh,
                      long long sWb, long long sWh,
                      long long sCb, long long sCh)
{
  constexpr int LDP = 40;                       // halfs per LDS row (80B pitch)
  __shared__ alignas(16) _Float16 As[64][LDP];
  __shared__ alignas(16) _Float16 Ws[64][LDP];

  const int z  = blockIdx.z;
  const int zb = z / zdiv, zh = z % zdiv;
  A += zb * sAb + zh * sAh;
  W += zb * sWb + zh * sWh;
  C += zb * sCb + zh * sCh;

  const int m0 = blockIdx.y * 64;
  const int n0 = blockIdx.x * 64;
  const int tid  = threadIdx.x;
  const int lane = tid & 31;
  const int wave = tid >> 5;

  // staging map: 64x32 tile = 512 float4 groups; 4 groups per thread
  int r4[4], c4[4];
  size_t aoff[4], woff[4];
  #pragma unroll
  for (int j = 0; j < 4; ++j) {
    int g = j * 128 + tid;
    r4[j] = g >> 3;            // row 0..63
    c4[j] = (g & 7) * 4;       // col 0,4,..,28
    aoff[j] = (size_t)(m0 + r4[j]) * ldaM + c4[j];
    woff[j] = (size_t)(n0 + r4[j]) * ldwN + (WVEC ? (size_t)c4[j] : (size_t)c4[j] * ldwK);
  }

  float4 va[4], vw[4];
  // issue all 8 tile loads back-to-back (global_load_b128), no consumers between
  auto gload = [&](int k0) {
    #pragma unroll
    for (int j = 0; j < 4; ++j)
      va[j] = *(const float4*)(A + aoff[j] + k0);
    #pragma unroll
    for (int j = 0; j < 4; ++j) {
      if constexpr (WVEC) {
        vw[j] = *(const float4*)(W + woff[j] + k0);
      } else {                                   // strided-K operand (P@V)
        const float* wpp = W + woff[j] + (size_t)k0 * ldwK;
        vw[j].x = wpp[0];
        vw[j].y = wpp[(size_t)ldwK];
        vw[j].z = wpp[(size_t)2 * ldwK];
        vw[j].w = wpp[(size_t)3 * ldwK];
      }
    }
  };
  auto stolds = [&]() {
    #pragma unroll
    for (int j = 0; j < 4; ++j) {
      v4h ha = { (_Float16)va[j].x, (_Float16)va[j].y, (_Float16)va[j].z, (_Float16)va[j].w };
      *(v4h*)&As[r4[j]][c4[j]] = ha;             // ds_store_b64
      v4h hw = { (_Float16)vw[j].x, (_Float16)vw[j].y, (_Float16)vw[j].z, (_Float16)vw[j].w };
      *(v4h*)&Ws[r4[j]][c4[j]] = hw;
    }
  };

  v8f acc[4] = {};
  const int arow = wave * 16 + (lane & 15);
  const int kb   = (lane >> 4) * 8;        // A half-lane K base (ISA 16-bit A layout)
  const int brow = lane & 15;              // B column N
  const int bko  = (lane >> 4) * 8;        // B half-lane K base (ISA 16-bit B layout)

  gload(0);                                 // prologue
  for (int k0 = 0; k0 < K; k0 += 32) {
    stolds();
    __syncthreads();
    if (k0 + 32 < K) gload(k0 + 32);        // overlap next-tile VMEM with WMMA
    if (k0 + 64 < K)                        // global_prefetch_b8 one tile further
      __builtin_prefetch(A + aoff[0] + k0 + 64, 0, 1);

    // A fragment: two 16B LDS loads -> v16h (K = {kb..kb+7, 16+kb..16+kb+7})
    v8h alo = *(const v8h*)&As[arow][kb];
    v8h ahi = *(const v8h*)&As[arow][16 + kb];
    v16h af = __builtin_shufflevector(alo, ahi, 0, 1, 2, 3, 4, 5, 6, 7,
                                      8, 9, 10, 11, 12, 13, 14, 15);
    #pragma unroll
    for (int nt = 0; nt < 4; ++nt) {
      v8h blo = *(const v8h*)&Ws[nt * 16 + brow][bko];
      v8h bhi = *(const v8h*)&Ws[nt * 16 + brow][bko + 16];
      v16h bf = __builtin_shufflevector(blo, bhi, 0, 1, 2, 3, 4, 5, 6, 7,
                                        8, 9, 10, 11, 12, 13, 14, 15);
      acc[nt] = __builtin_amdgcn_wmma_f32_16x16x32_f16(
          false, af, false, bf, (short)0, acc[nt], false, false);
    }
    __syncthreads();
  }

  // epilogue: C/D layout -> n = lane&15, VGPR r = M row (lane>=16 adds 8)
  const int cn  = lane & 15;
  const int mhi = (lane >> 4) * 8;
  #pragma unroll
  for (int nt = 0; nt < 4; ++nt) {
    #pragma unroll
    for (int r = 0; r < 8; ++r) {
      int gm = m0 + wave * 16 + mhi + r;
      int gn = n0 + nt * 16 + cn;
      float v = acc[nt][r] * alpha;
      if (bias) v += bias[gn];
      if      (act == 1) v = fmaxf(v, 0.f);                       // relu
      else if (act == 2) v = v / (1.f + __expf(-v));              // silu
      else if (act == 3) v = (v > 20.f) ? v : log1pf(__expf(v));  // softplus
      C[(size_t)gm * ldc + gn] = v;
    }
  }
}

// ============== softmax over rows of length 512 (wave per row) ==============
__global__ __launch_bounds__(256)
void softmax_rows_kernel(float* __restrict__ s)
{
  const int lane = threadIdx.x & 31;
  const int wave = threadIdx.x >> 5;
  const size_t row = (size_t)blockIdx.x * 8 + wave;  // B*H*L rows
  float* p = s + row * Ls;

  float vals[16];
  float mx = -1e30f;
  #pragma unroll
  for (int i = 0; i < 16; ++i) { vals[i] = p[lane + i * 32]; mx = fmaxf(mx, vals[i]); }
  #pragma unroll
  for (int off = 16; off; off >>= 1) mx = fmaxf(mx, __shfl_xor(mx, off, 32));
  float sum = 0.f;
  #pragma unroll
  for (int i = 0; i < 16; ++i) { vals[i] = __expf(vals[i] - mx); sum += vals[i]; }
  #pragma unroll
  for (int off = 16; off; off >>= 1) sum += __shfl_xor(sum, off, 32);
  const float inv = 1.f / sum;
  #pragma unroll
  for (int i = 0; i < 16; ++i) p[lane + i * 32] = vals[i] * inv;
}

// ====== causal depthwise conv1d (DC=4) + silu over (B,L,DI), x from xz ======
__global__ __launch_bounds__(256)
void conv_silu_kernel(const float* __restrict__ xz, const float* __restrict__ cw,
                      const float* __restrict__ cb, float* __restrict__ xc)
{
  size_t idx = (size_t)blockIdx.x * blockDim.x + threadIdx.x;  // B*L*DI
  int c = (int)(idx % DIi);
  int l = (int)((idx / DIi) % Ls);
  int b = (int)(idx / ((size_t)DIi * Ls));
  float acc = cb[c];
  #pragma unroll
  for (int k = 0; k < DCc; ++k) {
    int t = l + k - (DCc - 1);
    if (t >= 0)
      acc += xz[((size_t)(b * Ls + t)) * (2 * DIi) + c] * cw[c * DCc + k];
  }
  xc[idx] = acc / (1.f + __expf(-acc));   // silu
}

// ====== selective scan: one wave per (b,channel), 2 states per lane =========
__global__ __launch_bounds__(256)
void mamba_scan_kernel(const float* __restrict__ xc, const float* __restrict__ delta,
                       const float* __restrict__ proj, const float* __restrict__ xz,
                       const float* __restrict__ A_log, const float* __restrict__ Dp,
                       float* __restrict__ u)
{
  const int lane = threadIdx.x & 31;
  const int wave = threadIdx.x >> 5;
  const int ch = blockIdx.x * 8 + wave;   // [0, B*DI)
  const int b = ch / DIi, c = ch % DIi;
  const int s0 = lane, s1 = lane + 32;

  const float A0 = -__expf(A_log[c * DSs + s0]);
  const float A1 = -__expf(A_log[c * DSs + s1]);
  const float dcoef = Dp[c];
  const float* projb = proj + (size_t)b * Ls * (DTRr + 2 * DSs);

  float h0 = 0.f, h1 = 0.f;
  for (int t = 0; t < Ls; ++t) {
    const size_t rbase = (size_t)(b * Ls + t);
    const float dt = delta[rbase * DIi + c];
    const float x  = xc[rbase * DIi + c];
    const float Bv0 = projb[t * 192 + 64 + s0];
    const float Bv1 = projb[t * 192 + 64 + s1];
    const float Cv0 = projb[t * 192 + 128 + s0];
    const float Cv1 = projb[t * 192 + 128 + s1];
    h0 = h0 * __expf(dt * A0) + dt * x * Bv0;
    h1 = h1 * __expf(dt * A1) + dt * x * Bv1;
    float p = h0 * Cv0 + h1 * Cv1;
    #pragma unroll
    for (int off = 16; off; off >>= 1) p += __shfl_xor(p, off, 32);
    if (lane == 0) {
      const float y = p + x * dcoef;
      const float zv = xz[rbase * (2 * DIi) + DIi + c];
      const float sz = zv / (1.f + __expf(-zv));
      u[rbase * DIi + c] = y * sz;
    }
  }
}

// ========================= elementwise add ==================================
__global__ __launch_bounds__(256)
void add_kernel(const float* __restrict__ a, const float* __restrict__ b,
                float* __restrict__ o, int n)
{
  int i = blockIdx.x * blockDim.x + threadIdx.x;
  if (i < n) o[i] = a[i] + b[i];
}

// ================== layernorm over D=1024, block per row ====================
__global__ __launch_bounds__(256)
void layernorm_kernel(const float* __restrict__ x, const float* __restrict__ g,
                      const float* __restrict__ bta, float* __restrict__ out)
{
  const int row = blockIdx.x;           // B*L rows
  const float* xr = x + (size_t)row * Dd;
  __shared__ float sbuf[8];
  __shared__ float stat[2];
  const int tid = threadIdx.x, lane = tid & 31, wave = tid >> 5;

  float v[4];
  float s = 0.f;
  #pragma unroll
  for (int i = 0; i < 4; ++i) { v[i] = xr[tid + i * 256]; s += v[i]; }
  #pragma unroll
  for (int off = 16; off; off >>= 1) s += __shfl_xor(s, off, 32);
  if (lane == 0) sbuf[wave] = s;
  __syncthreads();
  if (tid == 0) { float t = 0.f; for (int i = 0; i < 8; ++i) t += sbuf[i]; stat[0] = t / Dd; }
  __syncthreads();
  const float mu = stat[0];

  float vs = 0.f;
  #pragma unroll
  for (int i = 0; i < 4; ++i) { float d = v[i] - mu; vs += d * d; }
  #pragma unroll
  for (int off = 16; off; off >>= 1) vs += __shfl_xor(vs, off, 32);
  if (lane == 0) sbuf[wave] = vs;
  __syncthreads();
  if (tid == 0) { float t = 0.f; for (int i = 0; i < 8; ++i) t += sbuf[i]; stat[1] = t / Dd; }
  __syncthreads();
  const float rstd = rsqrtf(stat[1] + EPSf);

  #pragma unroll
  for (int i = 0; i < 4; ++i) {
    int col = tid + i * 256;
    out[(size_t)row * Dd + col] = (v[i] - mu) * rstd * g[col] + bta[col];
  }
}

// =====================================================================
// host-side launcher
// =====================================================================
static inline void launch_gemm(hipStream_t stream,
                               const float* A, int ldaM,
                               const float* W, int ldwN, int ldwK,
                               const float* bias, float* C, int ldc,
                               int M, int N, int K, int act, float alpha,
                               int gz = 1, int zdiv = 1,
                               long long sAb = 0, long long sAh = 0,
                               long long sWb = 0, long long sWh = 0,
                               long long sCb = 0, long long sCh = 0)
{
  dim3 grid(N / 64, M / 64, gz), block(128);
  if (ldwK == 1)
    gemm_wmma_kernel<true><<<grid, block, 0, stream>>>(A, ldaM, W, ldwN, ldwK, bias, C, ldc,
                                                       K, act, alpha, zdiv,
                                                       sAb, sAh, sWb, sWh, sCb, sCh);
  else
    gemm_wmma_kernel<false><<<grid, block, 0, stream>>>(A, ldaM, W, ldwN, ldwK, bias, C, ldc,
                                                        K, act, alpha, zdiv,
                                                        sAb, sAh, sWb, sWh, sCb, sCh);
}

extern "C" void kernel_launch(void* const* d_in, const int* in_sizes, int n_in,
                              void* d_out, int out_size, void* d_ws, size_t ws_size,
                              hipStream_t stream)
{
  (void)in_sizes; (void)n_in; (void)out_size; (void)ws_size;
  const float* src        = (const float*)d_in[0];
  const float* attn_in_w  = (const float*)d_in[1];
  const float* attn_in_b  = (const float*)d_in[2];
  const float* attn_out_w = (const float*)d_in[3];
  const float* attn_out_b = (const float*)d_in[4];
  const float* ffn_w1     = (const float*)d_in[5];
  const float* ffn_b1     = (const float*)d_in[6];
  const float* ffn_w2     = (const float*)d_in[7];
  const float* ffn_b2     = (const float*)d_in[8];
  const float* ln_g       = (const float*)d_in[9];
  const float* ln_b       = (const float*)d_in[10];
  const float* m_in_w     = (const float*)d_in[11];
  const float* m_conv_w   = (const float*)d_in[12];
  const float* m_conv_b   = (const float*)d_in[13];
  const float* m_x_w      = (const float*)d_in[14];
  const float* m_dt_w     = (const float*)d_in[15];
  const float* m_dt_b     = (const float*)d_in[16];
  const float* m_A_log    = (const float*)d_in[17];
  const float* m_D        = (const float*)d_in[18];
  const float* m_out_w    = (const float*)d_in[19];
  float* out = (float*)d_out;
  float* ws  = (float*)d_ws;

  // ---- workspace layout (elements), overlapped by liveness ----
  const size_t SZ_XZ     = (size_t)MR * 2 * DIi;       // 4,194,304
  const size_t SZ_QKV    = (size_t)MR * 3 * Dd;        // 3,145,728
  const size_t SZ_SCORES = (size_t)Bb * Hh * Ls * Ls;  // 8,388,608
  const size_t SZ_BLD    = (size_t)MR * Dd;            // 1,048,576
  const size_t SZ_BLDI   = (size_t)MR * DIi;           // 2,097,152

  float* xz     = ws;                                  // [step2 .. scan]
  float* qkv    = xz + SZ_XZ;                          // [step1 .. AV]
  float* scores = qkv + SZ_QKV;                        // [score .. AV]
  float* bufD   = scores + SZ_SCORES;                  // attn_o -> mam_f -> ffo
  float* attn_f = bufD + SZ_BLD;                       // [attn proj .. add]
  // reuse dead regions:
  float* xc     = qkv;                                 // after AV
  float* projb  = qkv + SZ_BLDI;                       // fits inside qkv region
  float* delta  = scores;                              // after AV
  float* u      = scores + SZ_BLDI;
  float* hsum   = scores + 2 * SZ_BLDI;
  float* ffh    = xz;                                  // after scan
  float* attn_o = bufD;
  float* mam_f  = bufD;
  float* ffo    = bufD;

  const long long LL = Ls, DD = Dd;
  const long long sQKVb = LL * 3 * DD;                 // batch stride in qkv

  // 1) qkv = src @ attn_in_w.T + b          (1024 x 3072, K=1024)
  launch_gemm(stream, src, Dd, attn_in_w, Dd, 1, attn_in_b, qkv, 3 * Dd,
              MR, 3 * Dd, Dd, 0, 1.f);
  // 2) xz = src @ m_in_w.T                  (1024 x 4096, K=1024)
  launch_gemm(stream, src, Dd, m_in_w, Dd, 1, nullptr, xz, 2 * DIi,
              MR, 2 * DIi, Dd, 0, 1.f);
  // 3) scores[b,h] = Q Kt / 8               (512 x 512, K=64, z = b*H+h)
  launch_gemm(stream, qkv, 3 * Dd, qkv + Dd, 3 * Dd, 1, nullptr, scores, Ls,
              Ls, Ls, HDd, 0, 0.125f,
              Bb * Hh, Hh,
              sQKVb, HDd, sQKVb, HDd,
              (long long)Hh * Ls * Ls, (long long)Ls * Ls);
  // 4) softmax over rows
  softmax_rows_kernel<<<(Bb * Hh * Ls) / 8, 256, 0, stream>>>(scores);
  // 5) attn_o[b,:,h*64+ :] = P @ V          (512 x 64, K=512; W read transposed)
  launch_gemm(stream, scores, Ls, qkv + 2 * Dd, 1, 3 * Dd, nullptr, attn_o, Dd,
              Ls, HDd, Ls, 0, 1.f,
              Bb * Hh, Hh,
              (long long)Hh * Ls * Ls, (long long)Ls * Ls,
              sQKVb, HDd,
              LL * DD, HDd);
  // 6) attn_f = attn_o @ attn_out_w.T + b   (1024 x 1024, K=1024)
  launch_gemm(stream, attn_o, Dd, attn_out_w, Dd, 1, attn_out_b, attn_f, Dd,
              MR, Dd, Dd, 0, 1.f);
  // 7) xc = silu(causal_conv(xm))           elementwise over B*L*DI
  conv_silu_kernel<<<((size_t)MR * DIi) / 256, 256, 0, stream>>>(xz, m_conv_w, m_conv_b, xc);
  // 8) proj = xc @ m_x_w.T                  (1024 x 192, K=2048)
  launch_gemm(stream, xc, DIi, m_x_w, DIi, 1, nullptr, projb, DTRr + 2 * DSs,
              MR, DTRr + 2 * DSs, DIi, 0, 1.f);
  // 9) delta = softplus(proj[:, :64] @ m_dt_w.T + m_dt_b)   (1024 x 2048, K=64)
  launch_gemm(stream, projb, DTRr + 2 * DSs, m_dt_w, DTRr, 1, m_dt_b, delta, DIi,
              MR, DIi, DTRr, 3, 1.f);
  // 10) selective scan -> u = (y + xc*D) * silu(z)
  mamba_scan_kernel<<<(Bb * DIi) / 8, 256, 0, stream>>>(xc, delta, projb, xz, m_A_log, m_D, u);
  // 11) mam_f = u @ m_out_w.T               (1024 x 1024, K=2048)
  launch_gemm(stream, u, DIi, m_out_w, DIi, 1, nullptr, mam_f, Dd,
              MR, Dd, DIi, 0, 1.f);
  // 12) hsum = attn_f + mam_f
  add_kernel<<<(MR * Dd) / 256, 256, 0, stream>>>(attn_f, mam_f, hsum, MR * Dd);
  // 13) ffh = relu(hsum @ ffn_w1.T + b1)    (1024 x 4096, K=1024)
  launch_gemm(stream, hsum, Dd, ffn_w1, Dd, 1, ffn_b1, ffh, FFf,
              MR, FFf, Dd, 1, 1.f);
  // 14) ffo = ffh @ ffn_w2.T + b2           (1024 x 1024, K=4096)
  launch_gemm(stream, ffh, FFf, ffn_w2, FFf, 1, ffn_b2, ffo, Dd,
              MR, Dd, FFf, 0, 1.f);
  // 15) layernorm -> out
  layernorm_kernel<<<MR, 256, 0, stream>>>(ffo, ln_g, ln_b, out);
}
